// QuantDeepseekV3MLP_60601988546893
// MI455X (gfx1250) — 8-bit matrix operand order repaired and checked by reading, compiled, not run on hardware
//
#include <hip/hip_runtime.h>
#include <stdint.h>

typedef __attribute__((ext_vector_type(8))) int v8i;

#define MDIM 65536
#define HDIM 1024
#define IDIM 2816

#define LDA 80
#define LDB 80

__device__ __forceinline__ float sigmoid_f(float v) { return 1.0f / (1.0f + __expf(-v)); }

__device__ __forceinline__ int q8(float v, float inv) {
  int q = __float2int_rn(v * inv);
  q = q > 127 ? 127 : q;
  q = q < -128 ? -128 : q;
  return q;
}

__device__ __forceinline__ unsigned pack4(int a, int b, int c, int d) {
  return (unsigned)(a & 255) | ((unsigned)(b & 255) << 8) |
         ((unsigned)(c & 255) << 16) | ((unsigned)(d & 255) << 24);
}

__device__ __forceinline__ v8i wmma_iu8(v8i a, v8i b, v8i c) {
  return __builtin_amdgcn_wmma_i32_16x16x64_iu8(true, a, true, b, c, false, false);
}

union AFrag { v8i v; unsigned long long q[4]; };
union BFrag { v8i v; int4 x[2]; };

__device__ __forceinline__ unsigned lds_off(const void* p) {
  return (unsigned)(unsigned long long)p;
}

__device__ __forceinline__ void async_ld16(const char* g, unsigned l) {
  asm volatile("global_load_async_to_lds_b128 %0, %1, off" ::"v"(l), "v"(g) : "memory");
}
__device__ __forceinline__ void async_ld32(const char* g, unsigned l) {
  asm volatile(
      "global_load_async_to_lds_b128 %0, %1, off\n\t"
      "global_load_async_to_lds_b128 %0, %1, off offset:16" ::"v"(l), "v"(g)
      : "memory");
}
__device__ __forceinline__ void wait_async0() {
  asm volatile("s_wait_asynccnt 0x0" ::: "memory");
}

__global__ void init_kernel(unsigned* red) {
  if (threadIdx.x < 2) red[threadIdx.x] = 0u;
}

__global__ __launch_bounds__(256) void absmax_kernel(const float* __restrict__ x,
                                                     size_t n4,
                                                     unsigned* __restrict__ red) {
  __shared__ float s[256];
  float m = 0.f;
  const size_t stride = (size_t)gridDim.x * blockDim.x;
  for (size_t i = (size_t)blockIdx.x * blockDim.x + threadIdx.x; i < n4; i += stride) {
    float4 v = ((const float4*)x)[i];
    m = fmaxf(m, fmaxf(fmaxf(fabsf(v.x), fabsf(v.y)), fmaxf(fabsf(v.z), fabsf(v.w))));
  }
  s[threadIdx.x] = m;
  __syncthreads();
  for (int off = 128; off > 0; off >>= 1) {
    if (threadIdx.x < off) s[threadIdx.x] = fmaxf(s[threadIdx.x], s[threadIdx.x + off]);
    __syncthreads();
  }
  if (threadIdx.x == 0) atomicMax(red, __float_as_uint(s[0]));
}

__global__ void scale_kernel(const unsigned* __restrict__ red, int idx,
                             const float* __restrict__ clip,
                             float* __restrict__ scales) {
  if (threadIdx.x == 0)
    scales[idx] = sigmoid_f(clip[0]) * __uint_as_float(red[idx]) * (1.0f / 127.0f);
}

__global__ __launch_bounds__(256) void quant_act_kernel(const float* __restrict__ x,
                                                        size_t n4,
                                                        const float* __restrict__ scales,
                                                        int sidx,
                                                        unsigned* __restrict__ q) {
  const float s = scales[sidx];
  const float inv = s > 0.f ? 1.0f / s : 0.f;
  const size_t stride = (size_t)gridDim.x * blockDim.x;
  for (size_t i = (size_t)blockIdx.x * blockDim.x + threadIdx.x; i < n4; i += stride) {
    float4 v = ((const float4*)x)[i];
    q[i] = pack4(q8(v.x, inv), q8(v.y, inv), q8(v.z, inv), q8(v.w, inv));
  }
}

__global__ __launch_bounds__(256) void quant_w_kernel(const float* __restrict__ w,
                                                      int rowlen4,
                                                      char* __restrict__ wq,
                                                      float* __restrict__ srow) {
  __shared__ float s[256];
  const int row = blockIdx.x;
  const float4* wr = (const float4*)(w + (size_t)row * rowlen4 * 4);
  float m = 0.f;
  for (int i = threadIdx.x; i < rowlen4; i += 256) {
    float4 v = wr[i];
    m = fmaxf(m, fmaxf(fmaxf(fabsf(v.x), fabsf(v.y)), fmaxf(fabsf(v.z), fabsf(v.w))));
  }
  s[threadIdx.x] = m;
  __syncthreads();
  for (int off = 128; off > 0; off >>= 1) {
    if (threadIdx.x < off) s[threadIdx.x] = fmaxf(s[threadIdx.x], s[threadIdx.x + off]);
    __syncthreads();
  }
  const float scale = s[0] * (1.0f / 127.0f);
  const float inv = scale > 0.f ? 1.0f / scale : 0.f;
  unsigned* qr = (unsigned*)(wq + (size_t)row * rowlen4 * 4);
  for (int i = threadIdx.x; i < rowlen4; i += 256) {
    float4 v = wr[i];
    qr[i] = pack4(q8(v.x, inv), q8(v.y, inv), q8(v.z, inv), q8(v.w, inv));
  }
  if (threadIdx.x == 0) srow[row] = scale;
}

__device__ __forceinline__ void gemm1_compute(const char* sAb, const char* sBgb,
                                              const char* sBub, int wm, int wn, int lr,
                                              int kh, v8i (&accg)[2][2],
                                              v8i (&accu)[2][2]) {
  AFrag af[2];
#pragma unroll
  for (int mi = 0; mi < 2; ++mi) {
    const char* ab = sAb + (wm * 32 + mi * 16 + lr) * LDA + kh * 16;
#pragma unroll
    for (int g = 0; g < 4; ++g)
      af[mi].q[g] = *(const unsigned long long*)(ab + (g & 1) * 8 + (g >> 1) * 32);
  }
  BFrag bg[2], bu[2];
#pragma unroll
  for (int ni = 0; ni < 2; ++ni) {
    const char* cg = sBgb + (wn * 32 + ni * 16 + lr) * LDB + kh * 16;
    bg[ni].x[0] = *(const int4*)(cg);
    bg[ni].x[1] = *(const int4*)(cg + 32);
    const char* cu = sBub + (wn * 32 + ni * 16 + lr) * LDB + kh * 16;
    bu[ni].x[0] = *(const int4*)(cu);
    bu[ni].x[1] = *(const int4*)(cu + 32);
  }
#pragma unroll
  for (int mi = 0; mi < 2; ++mi)
#pragma unroll
    for (int ni = 0; ni < 2; ++ni) {
      accg[mi][ni] = wmma_iu8(af[mi].v, bg[ni].v, accg[mi][ni]);
      accu[mi][ni] = wmma_iu8(af[mi].v, bu[ni].v, accu[mi][ni]);
    }
}

__device__ __forceinline__ void gemm2_compute(const char* sAb, const char* sBb, int wm,
                                              int wn, int lr, int kh, v8i (&acc)[2][2]) {
  AFrag af[2];
#pragma unroll
  for (int mi = 0; mi < 2; ++mi) {
    const char* ab = sAb + (wm * 32 + mi * 16 + lr) * LDA + kh * 16;
#pragma unroll
    for (int g = 0; g < 4; ++g)
      af[mi].q[g] = *(const unsigned long long*)(ab + (g & 1) * 8 + (g >> 1) * 32);
  }
  BFrag bf[2];
#pragma unroll
  for (int ni = 0; ni < 2; ++ni) {
    const char* cb = sBb + (wn * 32 + ni * 16 + lr) * LDB + kh * 16;
    bf[ni].x[0] = *(const int4*)(cb);
    bf[ni].x[1] = *(const int4*)(cb + 32);
  }
#pragma unroll
  for (int mi = 0; mi < 2; ++mi)
#pragma unroll
    for (int ni = 0; ni < 2; ++ni)
      acc[mi][ni] = wmma_iu8(af[mi].v, bf[ni].v, acc[mi][ni]);
}

__global__ __launch_bounds__(256) void gemm1_kernel(
    const char* __restrict__ xq, const char* __restrict__ wqg,
    const char* __restrict__ wqu, const float* __restrict__ sg,
    const float* __restrict__ su, const float* __restrict__ scales,
    float* __restrict__ t, unsigned* __restrict__ red) {
  __shared__ __attribute__((aligned(16))) char sA[2][128 * LDA];
  __shared__ __attribute__((aligned(16))) char sBg[2][64 * LDB];
  __shared__ __attribute__((aligned(16))) char sBu[2][64 * LDB];
  __shared__ float sred[256];

  const int tid = threadIdx.x;
  const int wave = tid >> 5;
  const int lane = tid & 31;
  const int wm = wave & 3;
  const int wn = wave >> 2;
  const int lr = lane & 15;
  const int kh = lane >> 4;

  const int m0 = blockIdx.y * 128;
  const int n0 = blockIdx.x * 64;

  v8i accg[2][2], accu[2][2];
  {
    v8i z = {0, 0, 0, 0, 0, 0, 0, 0};
#pragma unroll
    for (int i = 0; i < 2; ++i)
#pragma unroll
      for (int j = 0; j < 2; ++j) { accg[i][j] = z; accu[i][j] = z; }
  }

  const int arow = tid >> 1, ahalf = tid & 1;
  const int btid = tid & 127;
  const int brow = btid >> 1, bhalf = btid & 1;
  const char* bsrcBase = (tid < 128) ? wqg : wqu;

  const char* aG = xq + (size_t)(m0 + arow) * HDIM + ahalf * 32;
  const char* bG = bsrcBase + (size_t)(n0 + brow) * HDIM + bhalf * 32;
  unsigned aL[2], bL[2];
  aL[0] = lds_off(&sA[0][arow * LDA + ahalf * 32]);
  aL[1] = lds_off(&sA[1][arow * LDA + ahalf * 32]);
  {
    const char* b0 = (tid < 128) ? &sBg[0][0] : &sBu[0][0];
    const char* b1 = (tid < 128) ? &sBg[1][0] : &sBu[1][0];
    bL[0] = lds_off(b0 + brow * LDB + bhalf * 32);
    bL[1] = lds_off(b1 + brow * LDB + bhalf * 32);
  }

  async_ld32(aG, aL[0]);
  async_ld32(bG, bL[0]);
  aG += 64;
  bG += 64;
  wait_async0();
  __syncthreads();

#pragma unroll 1
  for (int kt = 0; kt < HDIM / 128 - 1; ++kt) {
    async_ld32(aG, aL[1]);
    async_ld32(bG, bL[1]);
    aG += 64;
    bG += 64;
    gemm1_compute(sA[0], sBg[0], sBu[0], wm, wn, lr, kh, accg, accu);
    wait_async0();
    __syncthreads();
    async_ld32(aG, aL[0]);
    async_ld32(bG, bL[0]);
    aG += 64;
    bG += 64;
    gemm1_compute(sA[1], sBg[1], sBu[1], wm, wn, lr, kh, accg, accu);
    wait_async0();
    __syncthreads();
  }
  async_ld32(aG, aL[1]);
  async_ld32(bG, bL[1]);
  gemm1_compute(sA[0], sBg[0], sBu[0], wm, wn, lr, kh, accg, accu);
  wait_async0();
  __syncthreads();
  gemm1_compute(sA[1], sBg[1], sBu[1], wm, wn, lr, kh, accg, accu);

  const float s_x = scales[0];
  float lmax = 0.0f;
#pragma unroll
  for (int mi = 0; mi < 2; ++mi)
#pragma unroll
    for (int ni = 0; ni < 2; ++ni) {
      const int n = n0 + wn * 32 + ni * 16 + lr;
      const float sgn = s_x * sg[n];
      const float sun = s_x * su[n];
      const int mb = m0 + wm * 32 + mi * 16 + kh * 8;
#pragma unroll
      for (int r = 0; r < 8; ++r) {
        float g = sgn * (float)accg[mi][ni][r];
        float u = sun * (float)accu[mi][ni][r];
        float val = g * sigmoid_f(g) * u;
        t[(size_t)(mb + r) * IDIM + n] = val;
        lmax = fmaxf(lmax, fabsf(val));
      }
    }

  sred[tid] = lmax;
  __syncthreads();
  for (int off = 128; off > 0; off >>= 1) {
    if (tid < off) sred[tid] = fmaxf(sred[tid], sred[tid + off]);
    __syncthreads();
  }
  if (tid == 0) atomicMax(&red[1], __float_as_uint(sred[0]));
}

__global__ __launch_bounds__(256) void gemm2_kernel(
    const char* __restrict__ hq, const char* __restrict__ wqd,
    const float* __restrict__ sd, const float* __restrict__ scales,
    float* __restrict__ out) {
  __shared__ __attribute__((aligned(16))) char sA[2][128 * LDA];
  __shared__ __attribute__((aligned(16))) char sB[2][64 * LDB];

  const int tid = threadIdx.x;
  const int wave = tid >> 5;
  const int lane = tid & 31;
  const int wm = wave & 3;
  const int wn = wave >> 2;
  const int lr = lane & 15;
  const int kh = lane >> 4;

  const int m0 = blockIdx.y * 128;
  const int n0 = blockIdx.x * 64;

  v8i acc[2][2];
  {
    v8i z = {0, 0, 0, 0, 0, 0, 0, 0};
#pragma unroll
    for (int i = 0; i < 2; ++i)
#pragma unroll
      for (int j = 0; j < 2; ++j) acc[i][j] = z;
  }

  const int arow = tid >> 1, ahalf = tid & 1;
  const int b16row = tid >> 2, bq = tid & 3;

  const char* aG = hq + (size_t)(m0 + arow) * IDIM + ahalf * 32;
  const char* bG = wqd + (size_t)(n0 + b16row) * IDIM + bq * 16;
  unsigned aL[2], bL[2];
  aL[0] = lds_off(&sA[0][arow * LDA + ahalf * 32]);
  aL[1] = lds_off(&sA[1][arow * LDA + ahalf * 32]);
  bL[0] = lds_off(&sB[0][b16row * LDB + bq * 16]);
  bL[1] = lds_off(&sB[1][b16row * LDB + bq * 16]);

  async_ld32(aG, aL[0]);
  async_ld16(bG, bL[0]);
  aG += 64;
  bG += 64;
  wait_async0();
  __syncthreads();

#pragma unroll 1
  for (int kt = 0; kt < IDIM / 128 - 1; ++kt) {
    async_ld32(aG, aL[1]);
    async_ld16(bG, bL[1]);
    aG += 64;
    bG += 64;
    gemm2_compute(sA[0], sB[0], wm, wn, lr, kh, acc);
    wait_async0();
    __syncthreads();
    async_ld32(aG, aL[0]);
    async_ld16(bG, bL[0]);
    aG += 64;
    bG += 64;
    gemm2_compute(sA[1], sB[1], wm, wn, lr, kh, acc);
    wait_async0();
    __syncthreads();
  }
  async_ld32(aG, aL[1]);
  async_ld16(bG, bL[1]);
  gemm2_compute(sA[0], sB[0], wm, wn, lr, kh, acc);
  wait_async0();
  __syncthreads();
  gemm2_compute(sA[1], sB[1], wm, wn, lr, kh, acc);

  const float s_h = scales[1];
#pragma unroll
  for (int mi = 0; mi < 2; ++mi)
#pragma unroll
    for (int ni = 0; ni < 2; ++ni) {
      const int n = n0 + wn * 32 + ni * 16 + lr;
      const float sn = s_h * sd[n];
      const int mb = m0 + wm * 32 + mi * 16 + kh * 8;
#pragma unroll
      for (int r = 0; r < 8; ++r)
        out[(size_t)(mb + r) * HDIM + n] = sn * (float)acc[mi][ni][r];
    }
}

extern "C" void kernel_launch(void* const* d_in, const int* in_sizes, int n_in,
                              void* d_out, int out_size, void* d_ws, size_t ws_size,
                              hipStream_t stream) {
  (void)in_sizes; (void)n_in; (void)out_size; (void)ws_size;
  const float* x         = (const float*)d_in[0];
  const float* clip_x    = (const float*)d_in[1];
  const float* clip_down = (const float*)d_in[2];
  const float* w_gate    = (const float*)d_in[3];
  const float* w_up      = (const float*)d_in[4];
  const float* w_down    = (const float*)d_in[5];
  float* out = (float*)d_out;

  char* ws = (char*)d_ws;
  size_t off = 0;
  auto alloc = [&](size_t bytes) -> char* {
    char* p = ws + off;
    off = (off + bytes + 255) & ~(size_t)255;
    return p;
  };
  unsigned* red = (unsigned*)alloc(64);
  float* scales = (float*)alloc(64);
  float* sg = (float*)alloc((size_t)IDIM * 4);
  float* su = (float*)alloc((size_t)IDIM * 4);
  float* sd = (float*)alloc((size_t)HDIM * 4);
  char* xq  = alloc((size_t)MDIM * HDIM);
  char* wqg = alloc((size_t)IDIM * HDIM);
  char* wqu = alloc((size_t)IDIM * HDIM);
  char* wqd = alloc((size_t)HDIM * IDIM);
  char* hq  = alloc((size_t)MDIM * IDIM);
  float* t  = (float*)alloc((size_t)MDIM * IDIM * 4);

  init_kernel<<<1, 32, 0, stream>>>(red);
  absmax_kernel<<<2048, 256, 0, stream>>>(x, (size_t)MDIM * HDIM / 4, red);
  scale_kernel<<<1, 32, 0, stream>>>(red, 0, clip_x, scales);
  quant_act_kernel<<<2048, 256, 0, stream>>>(x, (size_t)MDIM * HDIM / 4, scales, 0,
                                             (unsigned*)xq);
  quant_w_kernel<<<IDIM, 256, 0, stream>>>(w_gate, HDIM / 4, wqg, sg);
  quant_w_kernel<<<IDIM, 256, 0, stream>>>(w_up, HDIM / 4, wqu, su);
  quant_w_kernel<<<HDIM, 256, 0, stream>>>(w_down, IDIM / 4, wqd, sd);
  gemm1_kernel<<<dim3(IDIM / 64, MDIM / 128), 256, 0, stream>>>(xq, wqg, wqu, sg, su,
                                                                scales, t, red);
  scale_kernel<<<1, 32, 0, stream>>>(red, 1, clip_down, scales);
  quant_act_kernel<<<4096, 256, 0, stream>>>(t, (size_t)MDIM * IDIM / 4, scales, 1,
                                             (unsigned*)hq);
  gemm2_kernel<<<dim3(HDIM / 64, MDIM / 128), 256, 0, stream>>>(hq, wqd, sd, scales, out);
}
